// ChildSumTreeLSTMCell_34574486733551
// MI455X (gfx1250) — matrix operand order repaired and checked by reading, not compiled, not run on hardware
//
#include <hip/hip_runtime.h>

// ---------------------------------------------------------------------------
// Child-Sum Tree-LSTM fused kernel for MI455X (gfx1250, wave32, WMMA).
// bf16 WMMA (v_wmma_f32_16x16x32_bf16, fp32 accumulate) for all seven
// x/h_sum GEMMs plus the per-child forget-gate GEMM, fused with gate math.
// LDS staging uses CDNA5 GLOBAL_LOAD_ASYNC_TO_LDS_B128 (ASYNCcnt) and the
// dominant per-child GEMM is double-buffered (copy/compute overlap).
// Workspace: 84 MiB of bf16 copies + transposed weights.
// ---------------------------------------------------------------------------

typedef __attribute__((ext_vector_type(16))) __bf16 v16bf;
typedef __attribute__((ext_vector_type(8)))  float  v8f;
typedef __bf16 bf16;

#define NN  8192
#define KCH 8
#define HH  512

union V8  { v8f v; float f[8]; };
union F16 { v16bf v; uint4 q[2]; };

#define WMMA_BF16(A, B, C) \
  __builtin_amdgcn_wmma_f32_16x16x32_bf16(false, (A), false, (B), (short)0, (C), false, false)

__device__ __forceinline__ float sigmoidf_(float x) { return 1.0f / (1.0f + __expf(-x)); }

// ---- CDNA5 async copy: 16 bytes per lane, global -> LDS, no VGPR staging ----
// LDS aperture is 4GiB-aligned, so the low 32 bits of a generic LDS pointer
// are the LDS byte offset the instruction wants in VDST.
__device__ __forceinline__ void async_cp16(const bf16* g, bf16* l) {
  unsigned lofs = (unsigned)(unsigned long long)l;
  asm volatile("global_load_async_to_lds_b128 %0, %1, off"
               :: "v"(lofs), "v"(g)
               : "memory");
}
__device__ __forceinline__ void wait_async0() {
  asm volatile("s_wait_asynccnt 0" ::: "memory");
}

// A-fragment (16x32 bf16, ISA 7.12.2): lane m = lane&15, hi = lane>>4.
// elements e=0..7 -> K = ko + 8*hi + e ; e=8..15 -> K = ko + 16 + 8*hi + (e-8).
template <int S>
__device__ __forceinline__ v16bf ldA(const bf16* s, int row, int hi, int ko) {
  F16 u;
  const bf16* p = s + row * S + ko + 8 * hi;
  u.q[0] = *(const uint4*)(p);
  u.q[1] = *(const uint4*)(p + 16);
  return u.v;
}

// B-fragment (32x16 bf16): lane col = lane&15, hi = lane>>4; same K order as ldA.
// Tiles are stored [col][k], so elements 0-7 and 8-15 are two 16-byte runs starting 32 bytes apart.
template <int S>
__device__ __forceinline__ v16bf ldB(const bf16* s, int col, int hi, int ko) {
  F16 u;
  const bf16* p = s + col * S + ko + 8 * hi;
  u.q[0] = *(const uint4*)(p);
  u.q[1] = *(const uint4*)(p + 16);
  return u.v;
}

// Cooperative async stage of a 64x32 bf16 tile (4 KiB): 256 threads x 16 B.
__device__ __forceinline__ void astage64x32(bf16* dst, const bf16* src, int stride, int tid) {
  const int r = tid >> 2;
  const int c = (tid & 3) << 3;
  async_cp16(src + (size_t)r * stride + c, dst + r * 32 + c);
}

// Cooperative async stage of a 64x64 bf16 tile (8 KiB): 256 threads x 32 B.
__device__ __forceinline__ void astage64x64(bf16* dst, const bf16* src, int stride, int tid) {
#pragma unroll
  for (int i = 0; i < 2; ++i) {
    const int id = tid + i * 256;
    const int r = id >> 3;
    const int c = (id & 7) << 3;
    async_cp16(src + (size_t)r * stride + c, dst + r * 64 + c);
  }
}

// ---------------------------------------------------------------------------
// Prep kernels
// ---------------------------------------------------------------------------
__global__ void prep_hsum(const float* __restrict__ x, const float* __restrict__ ch,
                          bf16* __restrict__ xb, bf16* __restrict__ hsb) {
  int i = blockIdx.x * 256 + threadIdx.x;     // < N*H (exact multiple)
  int n = i >> 9;
  int h = i & 511;
  const float* p = ch + (size_t)n * KCH * HH + h;
  float s = 0.f;
#pragma unroll
  for (int k = 0; k < KCH; ++k) s += p[(size_t)k * HH];
  hsb[i] = (bf16)s;
  xb[i]  = (bf16)x[i];
}

__global__ void cvt_bf16_k(const float* __restrict__ in, bf16* __restrict__ out, int total) {
  int i = blockIdx.x * 256 + threadIdx.x;
  if (i < total) out[i] = (bf16)in[i];
}

// Transpose + convert weights: dst[g][h][k] = (bf16) W_g[k][h].
__global__ void prep_wt(const float* W0, const float* W1, const float* W2, const float* W3,
                        const float* W4, const float* W5, const float* W6, const float* W7,
                        bf16* __restrict__ dst) {
  int g = blockIdx.y;
  const float* W;
  switch (g) {
    case 0: W = W0; break; case 1: W = W1; break;
    case 2: W = W2; break; case 3: W = W3; break;
    case 4: W = W4; break; case 5: W = W5; break;
    case 6: W = W6; break; default: W = W7; break;
  }
  int i = blockIdx.x * 256 + threadIdx.x;     // < 512*512 (exact multiple)
  int h = i >> 9;
  int k = i & 511;
  dst[(size_t)g * HH * HH + i] = (bf16)W[(size_t)k * HH + h];
}

// ---------------------------------------------------------------------------
// Fused main kernel: one 256-thread (8-wave) block per 64x64 output tile.
// wt holds 8 transposed bf16 matrices in order Wi,Ui,Wo,Uo,Wu,Uu,Wf,Uf.
// ---------------------------------------------------------------------------
__global__ __launch_bounds__(256)
void tlstm_fused(const bf16* __restrict__ xb, const bf16* __restrict__ hsb,
                 const bf16* __restrict__ chb, const float* __restrict__ child_c,
                 const bf16* __restrict__ wt,
                 const float* __restrict__ bWi, const float* __restrict__ bUi,
                 const float* __restrict__ bWf, const float* __restrict__ bUf,
                 const float* __restrict__ bWo, const float* __restrict__ bUo,
                 const float* __restrict__ bWu, const float* __restrict__ bUu,
                 float* __restrict__ out) {
  // 36 KiB arena.  Phase 1: sAx, sAh + 7 weight tiles (9 x 4 KiB, stride 32).
  // Phase 2: two (child 8 KiB + Uf 8 KiB) double buffers (32 KiB, stride 64).
  __shared__ alignas(16) bf16 smem[9 * 2048];

  const int tid  = threadIdx.x;
  const int lane = tid & 31;
  const int wave = tid >> 5;
  const int wm = wave & 3;            // 16-row subtile (4 groups cover 64 rows)
  const int wc = wave >> 2;           // 32-col half (2 groups cover 64 cols)
  const int hi = lane >> 4;
  const int ln = lane & 15;
  const int nbase = blockIdx.x << 6;
  const int hbase = blockIdx.y << 6;

  const int arow = (wm << 4) + ln;    // A-fragment row inside the 64-row tile
  const int cA = (wc << 5) + ln;      // column subtile 0 (within tile)
  const int cB = cA + 16;             // column subtile 1

  bf16* sAx = smem;
  bf16* sAh = smem + 2048;
#define SBT(g) (smem + (2 + (g)) * 2048)

  V8 ai0, ai1, ao0, ao1, au0, au1, af0, af1;
  ai0.v = (v8f)0.f; ai1.v = (v8f)0.f;
  ao0.v = (v8f)0.f; ao1.v = (v8f)0.f;
  au0.v = (v8f)0.f; au1.v = (v8f)0.f;
  af0.v = (v8f)0.f; af1.v = (v8f)0.f;

  // ---- Phase 1: i, o, u gates (x@W + h_sum@U) and x@Wf, reduced over 512 ----
#pragma unroll 1
  for (int kk = 0; kk < HH; kk += 32) {
    astage64x32(sAx, xb + (size_t)nbase * HH + kk, HH, tid);
    astage64x32(sAh, hsb + (size_t)nbase * HH + kk, HH, tid);
#pragma unroll
    for (int g = 0; g < 7; ++g)
      astage64x32(SBT(g), wt + (size_t)g * HH * HH + (size_t)hbase * HH + kk, HH, tid);
    wait_async0();
    __syncthreads();

    v16bf ax = ldA<32>(sAx, arow, hi, 0);
    v16bf ah = ldA<32>(sAh, arow, hi, 0);

    ai0.v = WMMA_BF16(ax, ldB<32>(SBT(0), cA, hi, 0), ai0.v);
    ai0.v = WMMA_BF16(ah, ldB<32>(SBT(1), cA, hi, 0), ai0.v);
    ai1.v = WMMA_BF16(ax, ldB<32>(SBT(0), cB, hi, 0), ai1.v);
    ai1.v = WMMA_BF16(ah, ldB<32>(SBT(1), cB, hi, 0), ai1.v);

    ao0.v = WMMA_BF16(ax, ldB<32>(SBT(2), cA, hi, 0), ao0.v);
    ao0.v = WMMA_BF16(ah, ldB<32>(SBT(3), cA, hi, 0), ao0.v);
    ao1.v = WMMA_BF16(ax, ldB<32>(SBT(2), cB, hi, 0), ao1.v);
    ao1.v = WMMA_BF16(ah, ldB<32>(SBT(3), cB, hi, 0), ao1.v);

    au0.v = WMMA_BF16(ax, ldB<32>(SBT(4), cA, hi, 0), au0.v);
    au0.v = WMMA_BF16(ah, ldB<32>(SBT(5), cA, hi, 0), au0.v);
    au1.v = WMMA_BF16(ax, ldB<32>(SBT(4), cB, hi, 0), au1.v);
    au1.v = WMMA_BF16(ah, ldB<32>(SBT(5), cB, hi, 0), au1.v);

    af0.v = WMMA_BF16(ax, ldB<32>(SBT(6), cA, hi, 0), af0.v);
    af1.v = WMMA_BF16(ax, ldB<32>(SBT(6), cB, hi, 0), af1.v);
    __syncthreads();
  }

  const int gcA = hbase + cA;
  const int gcB = hbase + cB;
  const float biA = bWi[gcA] + bUi[gcA], biB = bWi[gcB] + bUi[gcB];
  const float boA = bWo[gcA] + bUo[gcA], boB = bWo[gcB] + bUo[gcB];
  const float buA = bWu[gcA] + bUu[gcA], buB = bWu[gcB] + bUu[gcB];
  const float bfA = bWf[gcA] + bUf[gcA], bfB = bWf[gcB] + bUf[gcB];

  V8 c0, c1;
#pragma unroll
  for (int j = 0; j < 8; ++j) {
    c0.f[j] = sigmoidf_(ai0.f[j] + biA) * tanhf(au0.f[j] + buA);
    c1.f[j] = sigmoidf_(ai1.f[j] + biB) * tanhf(au1.f[j] + buB);
  }

  // ---- Phase 2: per-child forget gates f_k = sigmoid(x@Wf + child_h_k@Uf),
  //      c += f_k * child_c_k.  64-deep K chunks, double-buffered LDS.
  //      Ping-pong buffers are derived arithmetically from the parity bit
  //      (no LDS-pointer arrays: those become illegal static initializers). ----
  const bf16* ufsrc = wt + (size_t)7 * HH * HH + (size_t)hbase * HH;

#pragma unroll 1
  for (int k = 0; k < KCH; ++k) {
    V8 f0, f1;
    f0.v = af0.v;
    f1.v = af1.v;
    const bf16* asrc = chb + ((size_t)nbase * KCH + k) * HH;

    astage64x64(smem, asrc, KCH * HH, tid);          // child buffer 0
    astage64x64(smem + 4096, ufsrc, HH, tid);        // Uf buffer 0
    wait_async0();
    __syncthreads();

    int p = 0;
#pragma unroll 1
    for (int kk = 0; kk < HH; kk += 64) {
      bf16* chCur = smem + p * 8192;
      bf16* ufCur = chCur + 4096;
      if (kk + 64 < HH) {           // prefetch next chunk into the other buffer
        bf16* chNxt = smem + (p ^ 1) * 8192;
        astage64x64(chNxt, asrc + kk + 64, KCH * HH, tid);
        astage64x64(chNxt + 4096, ufsrc + kk + 64, HH, tid);
      }
      v16bf a0 = ldA<64>(chCur, arow, hi, 0);
      v16bf a1 = ldA<64>(chCur, arow, hi, 32);
      f0.v = WMMA_BF16(a0, ldB<64>(ufCur, cA, hi, 0), f0.v);
      f0.v = WMMA_BF16(a1, ldB<64>(ufCur, cA, hi, 32), f0.v);
      f1.v = WMMA_BF16(a0, ldB<64>(ufCur, cB, hi, 0), f1.v);
      f1.v = WMMA_BF16(a1, ldB<64>(ufCur, cB, hi, 32), f1.v);
      wait_async0();
      __syncthreads();
      p ^= 1;
    }

#pragma unroll
    for (int j = 0; j < 8; ++j) {
      int row = nbase + (wm << 4) + j + (hi << 3);
      const float* ccp = child_c + ((size_t)row * KCH + k) * HH;
      c0.f[j] += sigmoidf_(f0.f[j] + bfA) * ccp[gcA];
      c1.f[j] += sigmoidf_(f1.f[j] + bfB) * ccp[gcB];
    }
  }

  // ---- Epilogue: h = sigmoid(o) * tanh(c); write [h; c] planes ----
#pragma unroll
  for (int j = 0; j < 8; ++j) {
    int row = nbase + (wm << 4) + j + (hi << 3);
    float cv0 = c0.f[j], cv1 = c1.f[j];
    out[(size_t)row * HH + gcA] = sigmoidf_(ao0.f[j] + boA) * tanhf(cv0);
    out[(size_t)row * HH + gcB] = sigmoidf_(ao1.f[j] + boB) * tanhf(cv1);
    out[(size_t)NN * HH + (size_t)row * HH + gcA] = cv0;
    out[(size_t)NN * HH + (size_t)row * HH + gcB] = cv1;
  }
}

// ---------------------------------------------------------------------------
// Launch. Workspace layout (needs 84 MiB):
//   [0,8M)   x   bf16     [8M,16M)  h_sum bf16
//   [16M,80M) child_h bf16
//   [80M,84M) 8 transposed bf16 weight matrices (Wi,Ui,Wo,Uo,Wu,Uu,Wf,Uf)
// ---------------------------------------------------------------------------
extern "C" void kernel_launch(void* const* d_in, const int* in_sizes, int n_in,
                              void* d_out, int out_size, void* d_ws, size_t ws_size,
                              hipStream_t stream) {
  (void)in_sizes; (void)n_in; (void)out_size; (void)ws_size;

  const float* x   = (const float*)d_in[0];
  const float* chh = (const float*)d_in[1];
  const float* chc = (const float*)d_in[2];
  const float* Wi  = (const float*)d_in[3];  const float* bWi = (const float*)d_in[4];
  const float* Ui  = (const float*)d_in[5];  const float* bUi = (const float*)d_in[6];
  const float* Wf  = (const float*)d_in[7];  const float* bWf = (const float*)d_in[8];
  const float* Uf  = (const float*)d_in[9];  const float* bUf = (const float*)d_in[10];
  const float* Wo  = (const float*)d_in[11]; const float* bWo = (const float*)d_in[12];
  const float* Uo  = (const float*)d_in[13]; const float* bUo = (const float*)d_in[14];
  const float* Wu  = (const float*)d_in[15]; const float* bWu = (const float*)d_in[16];
  const float* Uu  = (const float*)d_in[17]; const float* bUu = (const float*)d_in[18];
  float* out = (float*)d_out;

  char* ws = (char*)d_ws;
  bf16* xb  = (bf16*)(ws);
  bf16* hsb = (bf16*)(ws + (size_t)8 * 1024 * 1024);
  bf16* chb = (bf16*)(ws + (size_t)16 * 1024 * 1024);
  bf16* wt  = (bf16*)(ws + (size_t)80 * 1024 * 1024);

  prep_hsum<<<(NN * HH) / 256, 256, 0, stream>>>(x, chh, xb, hsb);
  cvt_bf16_k<<<(NN * KCH * HH) / 256, 256, 0, stream>>>(chh, chb, NN * KCH * HH);
  prep_wt<<<dim3((HH * HH) / 256, 8), 256, 0, stream>>>(Wi, Ui, Wo, Uo, Wu, Uu, Wf, Uf, wt);

  tlstm_fused<<<dim3(NN / 64, HH / 64), 256, 0, stream>>>(
      xb, hsb, chb, chc, wt, bWi, bUi, bWf, bUf, bWo, bUo, bWu, bUu, out);
}
